// AttnBlock_77730318123634
// MI455X (gfx1250) — matrix operand order repaired and checked by reading, not compiled, not run on hardware
//
#include <hip/hip_runtime.h>
#include <hip/hip_bf16.h>

// ---------------------------------------------------------------------------
// AttnBlock for MI455X (gfx1250): GroupNorm -> QKV -> flash attention -> proj
// All GEMMs use v_wmma_f32_16x16x32_bf16 (bf16 in, f32 accumulate).
// Weights are pre-converted to bf16 once (wconv_kernel) so GEMM hot loops are
// pure {global_load_b128 / ds_load_b128 -> v_wmma} streams.
// Fragment layouts (wave32):
//   A-frag (16x32, lane=row M): lanes 0-15 hold K[0..15];
//                               lanes 16-31 hold K[16..31]
//   B-frag (32x16, lane=col N): lane%16=col, K-run of 16 contiguous,
//                               base K = (lane/16)*16
//   C/D   (16x16 f32): vgpr r, lane l -> (M = r + (l/16)*8, N = l%16)
// ---------------------------------------------------------------------------

typedef __attribute__((ext_vector_type(16))) __bf16 bf16x16;
typedef __attribute__((ext_vector_type(8)))  float  f32x8;
typedef __attribute__((ext_vector_type(4)))  int    i32x4;

#define CDIM 512
#define HWN  4096
#define NB   4
#define LOG2E 1.4426950408889634f

#if defined(__gfx1250__) && __has_builtin(__builtin_amdgcn_global_load_async_to_lds_b128)
#define HAVE_ASYNC_LDS 1
// pointee typedefs carrying explicit address spaces (AS1 = global/__device__,
// AS3 = LDS); C-style casts below lower to addrspacecast.
typedef __attribute__((address_space(1))) i32x4 gi32x4;
typedef __attribute__((address_space(3))) i32x4 li32x4;
#else
#define HAVE_ASYNC_LDS 0
#endif

static __device__ __forceinline__ __bf16 f2bf(float f) {
  unsigned u = __builtin_bit_cast(unsigned, f);
  unsigned r = (u + 0x7FFFu + ((u >> 16) & 1u)) >> 16;  // round-to-nearest-even
  unsigned short h = (unsigned short)r;
  return __builtin_bit_cast(__bf16, h);
}

// assemble a v16bf fragment from two 16-byte chunks
static __device__ __forceinline__ bf16x16 frag2(const __bf16* lo, const __bf16* hi) {
  union { uint4 u[2]; bf16x16 v; } t;
  t.u[0] = *(const uint4*)lo;
  t.u[1] = *(const uint4*)hi;
  return t.v;
}

static __device__ __forceinline__ f32x8 wmma_bf16(bf16x16 a, bf16x16 b, f32x8 c) {
  return __builtin_amdgcn_wmma_f32_16x16x32_bf16(false, a, false, b, (short)0, c,
                                                 false, false);
}

// A-fragment from a bf16 row-major (Cout x Cin) matrix. 'half' = lane/16.
static __device__ __forceinline__ bf16x16 load_a_bf(const __bf16* __restrict__ w,
                                                    int row, int c0, int half) {
  const __bf16* p = w + (size_t)row * CDIM + c0 + half * 16;
  return frag2(p, p + 8);
}

// ---------------------------------------------------------------------------
// Kernel 0: one-shot f32 -> bf16 conversion of the four 512x512 weights.
// ---------------------------------------------------------------------------
__global__ void __launch_bounds__(256)
wconv_kernel(const float* __restrict__ wq, const float* __restrict__ wk,
             const float* __restrict__ wv, const float* __restrict__ wo,
             __bf16* __restrict__ wb) {
  const int m = blockIdx.y;  // 0..3
  const float* src = (m == 0) ? wq : (m == 1) ? wk : (m == 2) ? wv : wo;
  __bf16* dst = wb + (size_t)m * CDIM * CDIM;
  const int i = (blockIdx.x * 256 + threadIdx.x) * 4;
  float4 v = *(const float4*)(src + i);
  dst[i + 0] = f2bf(v.x);
  dst[i + 1] = f2bf(v.y);
  dst[i + 2] = f2bf(v.z);
  dst[i + 3] = f2bf(v.w);
}

// ---------------------------------------------------------------------------
// Kernel 1: GroupNorm, writes hT (B, HW, C) bf16  (c-contiguous = B-operand)
// one block per (batch, group); 16 channels x 4096 px = 65536 elements
// ---------------------------------------------------------------------------
__global__ void __launch_bounds__(256)
gn_kernel(const float* __restrict__ x, const float* __restrict__ gw,
          const float* __restrict__ gb, __bf16* __restrict__ hT) {
  const int bg = blockIdx.x;
  const int b = bg >> 5;
  const int g = bg & 31;
  const int tid = threadIdx.x;
  const size_t xbase = ((size_t)b * CDIM + g * 16) * HWN;

  float s = 0.f, ss = 0.f;
  for (int i = tid; i < 16 * HWN; i += 256) {
    float v = x[xbase + i];
    s += v; ss += v * v;
  }
  #pragma unroll
  for (int off = 16; off; off >>= 1) {
    s  += __shfl_xor(s,  off, 32);
    ss += __shfl_xor(ss, off, 32);
  }
  __shared__ float rs[8], rq[8], smean, srstd;
  const int wid = tid >> 5, lane = tid & 31;
  if (lane == 0) { rs[wid] = s; rq[wid] = ss; }
  __syncthreads();
  if (tid == 0) {
    float S = 0.f, Q = 0.f;
    #pragma unroll
    for (int i = 0; i < 8; i++) { S += rs[i]; Q += rq[i]; }
    float mean = S * (1.f / 65536.f);
    float var  = Q * (1.f / 65536.f) - mean * mean;
    smean = mean;
    srstd = rsqrtf(var + 1e-5f);
  }
  __syncthreads();
  const float mean = smean, rstd = srstd;
  for (int i = tid; i < 16 * HWN; i += 256) {
    int cl = i >> 12, p = i & (HWN - 1);
    int c = g * 16 + cl;
    float v = (x[xbase + i] - mean) * rstd * gw[c] + gb[c];
    hT[((size_t)b * HWN + p) * CDIM + c] = f2bf(v);
  }
}

// ---------------------------------------------------------------------------
// Kernel 2: QKV projections.  out = W * h + bias
// grid (HW/128, C/128, 3*B); 8 waves, wave tile 32(M=outch) x 64(N=pixel)
// q -> qT (HW,C) pre-scaled by 1/sqrt(C);  k -> kT (HW,C);  v -> (C,HW)
// ---------------------------------------------------------------------------
__global__ void __launch_bounds__(256)
qkv_kernel(const __bf16* __restrict__ hT, const __bf16* __restrict__ wb,
           const float* __restrict__ bq, const float* __restrict__ bk,
           const float* __restrict__ bv,
           __bf16* __restrict__ qT, __bf16* __restrict__ kT,
           __bf16* __restrict__ vbuf) {
  const int p0 = blockIdx.x * 128;
  const int o0 = blockIdx.y * 128;
  const int which = blockIdx.z % 3;
  const int b = blockIdx.z / 3;
  const int tid = threadIdx.x;
  const int wave = tid >> 5, lane = tid & 31;
  const int half = lane >> 4, ln = lane & 15;
  const int mW = o0 + (wave & 3) * 32;
  const int nW = p0 + (wave >> 2) * 64;

  const __bf16* W  = wb + (size_t)which * CDIM * CDIM;
  const float*  Bv = (which == 0) ? bq : (which == 1) ? bk : bv;

  const f32x8 zero = {0.f, 0.f, 0.f, 0.f, 0.f, 0.f, 0.f, 0.f};
  f32x8 acc[2][4];
  #pragma unroll
  for (int i = 0; i < 2; i++)
    #pragma unroll
    for (int j = 0; j < 4; j++) acc[i][j] = zero;

  const __bf16* hbase = hT + (size_t)b * HWN * CDIM;

  for (int cs = 0; cs < CDIM; cs += 32) {
    bf16x16 afr[2];
    #pragma unroll
    for (int mt = 0; mt < 2; mt++)
      afr[mt] = load_a_bf(W, mW + mt * 16 + ln, cs, half);
    #pragma unroll
    for (int nt = 0; nt < 4; nt++) {
      const __bf16* bp =
          hbase + (size_t)(nW + nt * 16 + ln) * CDIM + cs + half * 16;
      bf16x16 bfr = frag2(bp, bp + 8);
      #pragma unroll
      for (int mt = 0; mt < 2; mt++)
        acc[mt][nt] = wmma_bf16(afr[mt], bfr, acc[mt][nt]);
    }
  }

  const float qs = 0.044194173824159216f;  // 1/sqrt(512)
  #pragma unroll
  for (int mt = 0; mt < 2; mt++)
    #pragma unroll
    for (int nt = 0; nt < 4; nt++)
      #pragma unroll
      for (int r = 0; r < 8; r++) {
        int o = mW + mt * 16 + r + half * 8;
        int p = nW + nt * 16 + ln;
        float vv = acc[mt][nt][r] + Bv[o];
        if (which == 0)
          qT[((size_t)b * HWN + p) * CDIM + o] = f2bf(vv * qs);
        else if (which == 1)
          kT[((size_t)b * HWN + p) * CDIM + o] = f2bf(vv);
        else
          vbuf[((size_t)b * CDIM + o) * HWN + p] = f2bf(vv);
      }
}

// ---------------------------------------------------------------------------
// Kernel 3: flash attention over 4096 keys. One block (4 waves) per 16-query
// tile. Each wave computes one 16x16 score slice per 64-key chunk, softmax
// stats shared through LDS, then accumulates O^T for its 128-channel slice.
// Q tile staged into LDS with GLOBAL_LOAD_ASYNC_TO_LDS when available.
// Writes oT (B, HW, C) bf16 (B-operand layout for the final projection).
// ---------------------------------------------------------------------------
__global__ void __launch_bounds__(128)
attn_kernel(const __bf16* __restrict__ qT, const __bf16* __restrict__ kT,
            const __bf16* __restrict__ vbuf, __bf16* __restrict__ oT) {
  const int qt = blockIdx.x;
  const int b  = blockIdx.y;
  const int qp0 = qt * 16;
  const int tid = threadIdx.x;
  const int w = tid >> 5, lane = tid & 31;
  const int half = lane >> 4, ln = lane & 15;

  __shared__ __align__(16) __bf16 sQ[16 * CDIM];  // 16 KB
  __shared__ __align__(16) __bf16 sP[16 * 64];    // 2 KB
  __shared__ float sM[16], sL[16], sA[16];
  __shared__ float pMax[4][16], pSum[4][16];

  // stage Q tile (16 rows x 512 ch) -> LDS, flat copy of 1024 x 16B
  {
    const __bf16* src = qT + ((size_t)b * HWN + qp0) * CDIM;
#if HAVE_ASYNC_LDS
    for (int i = tid; i < 1024; i += 128) {
      __builtin_amdgcn_global_load_async_to_lds_b128(
          (gi32x4*)(src + i * 8), (li32x4*)(sQ + i * 8), 0, 0);
    }
#if __has_builtin(__builtin_amdgcn_s_wait_asynccnt)
    __builtin_amdgcn_s_wait_asynccnt(0);
#else
    asm volatile("s_wait_asynccnt 0" ::: "memory");
#endif
#else
    const uint4* s4 = (const uint4*)src;
    uint4* d4 = (uint4*)sQ;
    for (int i = tid; i < 1024; i += 128) d4[i] = s4[i];
#endif
  }
  if (tid < 16) { sM[tid] = -1e30f; sL[tid] = 0.f; }
  __syncthreads();

  const f32x8 zero = {0.f, 0.f, 0.f, 0.f, 0.f, 0.f, 0.f, 0.f};
  f32x8 oacc[8];
  #pragma unroll
  for (int i = 0; i < 8; i++) oacc[i] = zero;

  const int cW = w * 128;  // this wave's channel slice for O

  for (int kc = 0; kc < HWN; kc += 64) {
    const int kp0 = kc + w * 16;

    // ---- scores S = Q^T K for this wave's 16x16 slice --------------------
    f32x8 s = zero;
    for (int cs = 0; cs < CDIM; cs += 32) {
      const __bf16* ap = sQ + (size_t)ln * CDIM + cs + half * 16;
      bf16x16 afr = frag2(ap, ap + 8);
      const __bf16* bp =
          kT + ((size_t)b * HWN + kp0 + ln) * CDIM + cs + half * 16;
      bf16x16 bfr = frag2(bp, bp + 8);
      s = wmma_bf16(afr, bfr, s);
    }

    // ---- per-row max across the 16 lanes of each half --------------------
    float rm[8];
    #pragma unroll
    for (int r = 0; r < 8; r++) rm[r] = s[r];
    #pragma unroll
    for (int off = 1; off < 16; off <<= 1)
      #pragma unroll
      for (int r = 0; r < 8; r++)
        rm[r] = fmaxf(rm[r], __shfl_xor(rm[r], off, 32));
    if (ln == 0)
      #pragma unroll
      for (int r = 0; r < 8; r++) pMax[w][r + half * 8] = rm[r];
    __syncthreads();

    if (tid < 16) {
      float mc = fmaxf(fmaxf(pMax[0][tid], pMax[1][tid]),
                       fmaxf(pMax[2][tid], pMax[3][tid]));
      float mold = sM[tid];
      float mnew = fmaxf(mold, mc);
      sA[tid] = exp2f((mold - mnew) * LOG2E);
      sM[tid] = mnew;
    }
    __syncthreads();

    // ---- P = exp(S - m); write A-layout bf16 tile to LDS; row sums -------
    float rsum[8];
    #pragma unroll
    for (int r = 0; r < 8; r++) {
      float m = sM[r + half * 8];
      float pv = exp2f((s[r] - m) * LOG2E);
      sP[(r + half * 8) * 64 + w * 16 + ln] = f2bf(pv);
      rsum[r] = pv;
    }
    #pragma unroll
    for (int off = 1; off < 16; off <<= 1)
      #pragma unroll
      for (int r = 0; r < 8; r++) rsum[r] += __shfl_xor(rsum[r], off, 32);
    if (ln == 0)
      #pragma unroll
      for (int r = 0; r < 8; r++) pSum[w][r + half * 8] = rsum[r];

    // rescale running O accumulators by alpha(row)
    #pragma unroll
    for (int r = 0; r < 8; r++) {
      float a = sA[r + half * 8];
      #pragma unroll
      for (int t = 0; t < 8; t++) oacc[t][r] *= a;
    }
    __syncthreads();

    if (tid < 16)
      sL[tid] = sL[tid] * sA[tid] +
                pSum[0][tid] + pSum[1][tid] + pSum[2][tid] + pSum[3][tid];

    // ---- O^T += P * V^T  (A from LDS, B straight from v (C,HW)) ----------
    #pragma unroll
    for (int kps = 0; kps < 64; kps += 32) {
      const __bf16* ap = sP + (size_t)ln * 64 + kps + half * 16;
      bf16x16 afr = frag2(ap, ap + 8);
      #pragma unroll
      for (int ct = 0; ct < 8; ct++) {
        int c = cW + ct * 16 + ln;
        const __bf16* bp =
            vbuf + ((size_t)b * CDIM + c) * HWN + kc + kps + half * 16;
        bf16x16 bfr = frag2(bp, bp + 8);
        oacc[ct] = wmma_bf16(afr, bfr, oacc[ct]);
      }
    }
    __syncthreads();
  }

  // ---- normalize and store oT (qp-major, c contiguous) -------------------
  #pragma unroll
  for (int r = 0; r < 8; r++) {
    int row = r + half * 8;
    float inv = 1.f / sL[row];
    #pragma unroll
    for (int ct = 0; ct < 8; ct++) {
      int c = cW + ct * 16 + ln;
      oT[((size_t)b * HWN + qp0 + row) * CDIM + c] = f2bf(oacc[ct][r] * inv);
    }
  }
}

// ---------------------------------------------------------------------------
// Kernel 4: out = x + Wo * o + bo   (same GEMM skeleton as QKV, f32 out)
// ---------------------------------------------------------------------------
__global__ void __launch_bounds__(256)
proj_kernel(const __bf16* __restrict__ oT, const __bf16* __restrict__ wob,
            const float* __restrict__ bo, const float* __restrict__ x,
             float* __restrict__ out) {
  const int p0 = blockIdx.x * 128;
  const int o0 = blockIdx.y * 128;
  const int b  = blockIdx.z;
  const int tid = threadIdx.x;
  const int wave = tid >> 5, lane = tid & 31;
  const int half = lane >> 4, ln = lane & 15;
  const int mW = o0 + (wave & 3) * 32;
  const int nW = p0 + (wave >> 2) * 64;

  const f32x8 zero = {0.f, 0.f, 0.f, 0.f, 0.f, 0.f, 0.f, 0.f};
  f32x8 acc[2][4];
  #pragma unroll
  for (int i = 0; i < 2; i++)
    #pragma unroll
    for (int j = 0; j < 4; j++) acc[i][j] = zero;

  const __bf16* obase = oT + (size_t)b * HWN * CDIM;

  for (int cs = 0; cs < CDIM; cs += 32) {
    bf16x16 afr[2];
    #pragma unroll
    for (int mt = 0; mt < 2; mt++)
      afr[mt] = load_a_bf(wob, mW + mt * 16 + ln, cs, half);
    #pragma unroll
    for (int nt = 0; nt < 4; nt++) {
      const __bf16* bp =
          obase + (size_t)(nW + nt * 16 + ln) * CDIM + cs + half * 16;
      bf16x16 bfr = frag2(bp, bp + 8);
      #pragma unroll
      for (int mt = 0; mt < 2; mt++)
        acc[mt][nt] = wmma_bf16(afr[mt], bfr, acc[mt][nt]);
    }
  }

  #pragma unroll
  for (int mt = 0; mt < 2; mt++)
    #pragma unroll
    for (int nt = 0; nt < 4; nt++)
      #pragma unroll
      for (int r = 0; r < 8; r++) {
        int o = mW + mt * 16 + r + half * 8;
        int p = nW + nt * 16 + ln;
        size_t idx = ((size_t)b * CDIM + o) * HWN + p;
        out[idx] = x[idx] + bo[o] + acc[mt][nt][r];
      }
}

// ---------------------------------------------------------------------------
extern "C" void kernel_launch(void* const* d_in, const int* in_sizes, int n_in,
                              void* d_out, int out_size, void* d_ws,
                              size_t ws_size, hipStream_t stream) {
  (void)in_sizes; (void)n_in; (void)out_size; (void)ws_size;
  const float* x    = (const float*)d_in[0];
  const float* gn_w = (const float*)d_in[1];
  const float* gn_b = (const float*)d_in[2];
  const float* wq   = (const float*)d_in[3];
  const float* bq   = (const float*)d_in[4];
  const float* wk   = (const float*)d_in[5];
  const float* bk   = (const float*)d_in[6];
  const float* wv   = (const float*)d_in[7];
  const float* bv   = (const float*)d_in[8];
  const float* wo   = (const float*)d_in[9];
  const float* bo   = (const float*)d_in[10];
  float* out = (float*)d_out;

  const size_t SZ = (size_t)NB * CDIM * HWN;  // elements per (B,C,HW) tensor
  __bf16* hT = (__bf16*)d_ws;     // (B, HW, C)
  __bf16* qT = hT + SZ;           // (B, HW, C), pre-scaled by 1/sqrt(C)
  __bf16* kT = qT + SZ;           // (B, HW, C)
  __bf16* vb = kT + SZ;           // (B, C, HW)
  __bf16* wb = vb + SZ;           // 4 x (512 x 512) bf16 weights
  __bf16* oT = hT;                // reuse hT (dead after QKV) for (B, HW, C)
  // total workspace: 64 MB + 2 MB

  wconv_kernel<<<dim3(CDIM * CDIM / 1024, 4), 256, 0, stream>>>(
      wq, wk, wv, wo, wb);
  gn_kernel<<<dim3(NB * 32), 256, 0, stream>>>(x, gn_w, gn_b, hT);
  qkv_kernel<<<dim3(HWN / 128, CDIM / 128, 3 * NB), 256, 0, stream>>>(
      hT, wb, bq, bk, bv, qT, kT, vb);
  attn_kernel<<<dim3(HWN / 16, NB), 128, 0, stream>>>(qT, kT, vb, oT);
  proj_kernel<<<dim3(HWN / 128, CDIM / 128, NB), 256, 0, stream>>>(
      oT, wb + (size_t)3 * CDIM * CDIM, bo, x, out);
}
